// NeuConV2Net_28948079575669
// MI455X (gfx1250) — matrix operand order repaired and checked by reading, not compiled, not run on hardware
//
#include <hip/hip_runtime.h>
#include <hip/hip_bf16.h>

// ---------------------------------------------------------------------------
// NeuConV2 tri-plane feature net for gfx1250 (MI455X).
//   1) pack_w_kernel      : f32 conv weights -> bf16 tiles in CDNA5 WMMA A-layout
//   2) bp_kernel          : multi-view back-projection + cosine-sim -> bf16 vol planes + f32 counts
//   3) conv_wmma          : implicit-GEMM 3x3 conv. Workgroup stages the WMMA-B
//                           operand for all K-chunks in LDS once (shared by all
//                           6 waves / all M-tiles), then each wave runs
//                           ds_load(B) + global(A) + v_wmma_f32_16x16x32_bf16
//                           with 1 (conv1) or 3 (conv2) accumulators per wave.
//   4) upsample_sum_kernel: per-plane linear upsample along downsampled axis, sum 3 planes
//   5) mask_kernel        : nearest-resampled counts, any(count>1)
// ---------------------------------------------------------------------------

typedef __attribute__((ext_vector_type(16))) __bf16 v16bf;
typedef __attribute__((ext_vector_type(8)))  float  v8f;

#define NVOX        96
#define PLANE_PTS   221184      // 24*96*96
#define SPATIAL     9216        // 96*96
#define C2D         24
#define HH          120
#define WW          160
#define NVIEW       9
#define TENSO_CH    12
#define FEAT_ELEMS  10616832    // 12*96*96*96
#define MASK_ELEMS  884736      // 96*96*96

// ------------------------------ weight packing -----------------------------
// A-matrix 16x32 bf16 layout (ISA 7.12.2): lane l (M = l&15), VGPR v, elem e:
//   k = ((v>=4)?16:0) + ((l>=16)?8:0) + (v&3)*2 + e
__global__ void pack_w_kernel(const float* __restrict__ w, __bf16* __restrict__ dst,
                              int Cout, int Cin, int numKC) {
    int total = (Cout / 16) * numKC * 32;
    int t = blockIdx.x * blockDim.x + threadIdx.x;
    if (t >= total) return;
    int K = Cin * 9;
    int lane = t & 31;
    int tile = t >> 5;               // mt*numKC + kc
    int kc = tile % numKC;
    int mt = tile / numKC;
    int m = mt * 16 + (lane & 15);
    int hi8 = (lane >= 16) ? 8 : 0;
#pragma unroll
    for (int j = 0; j < 16; ++j) {
        int v = j >> 1, e = j & 1;
        int k = kc * 32 + ((v >= 4) ? 16 : 0) + hi8 + (v & 3) * 2 + e;
        float val = 0.0f;
        if (k < K) {
            int ic = k / 9;
            int kk = k - ic * 9;
            val = w[(m * Cin + ic) * 9 + kk];
        }
        dst[(size_t)tile * 512 + lane * 16 + j] = (__bf16)val;
    }
}

// ------------------------------ back-projection ----------------------------
__device__ __forceinline__ void sample24(const float* __restrict__ fv,
                                         float x, float y, float vm,
                                         float* __restrict__ out) {
    float x0f = floorf(x);
    x0f = fminf(fmaxf(x0f, 0.0f), (float)(WW - 2));
    float y0f = floorf(y);
    y0f = fminf(fmaxf(y0f, 0.0f), (float)(HH - 2));
    float wx = x - x0f, wy = y - y0f;
    int base = (int)y0f * WW + (int)x0f;
    float w00 = (1.0f - wy) * (1.0f - wx);
    float w01 = (1.0f - wy) * wx;
    float w10 = wy * (1.0f - wx);
    float w11 = wy * wx;
#pragma unroll
    for (int c = 0; c < C2D; ++c) {
        const float* f = fv + c * (HH * WW);
        float v = f[base] * w00 + f[base + 1] * w01 +
                  f[base + WW] * w10 + f[base + WW + 1] * w11;
        out[c] = v * vm;
    }
}

__global__ void bp_kernel(const float* __restrict__ feat2d,
                          const float* __restrict__ KR,
                          const float* __restrict__ vol_origin,
                          __bf16* __restrict__ volBf,
                          float* __restrict__ cnt) {
    int id = blockIdx.x * blockDim.x + threadIdx.x;
    if (id >= 3 * PLANE_PTS) return;
    int p = id / PLANE_PTS;
    int r = id - p * PLANE_PTS;
    int c  = r / SPATIAL;
    int rem = r - c * SPATIAL;
    int yy = rem / NVOX, xx = rem - (rem / NVOX) * NVOX;

    float ix, iy, iz;
    if (p == 0)      { ix = 4.0f * c; iy = (float)yy; iz = (float)xx; }
    else if (p == 1) { ix = (float)yy; iy = 4.0f * c; iz = (float)xx; }
    else             { ix = (float)yy; iy = (float)xx; iz = 4.0f * c; }
    float px = vol_origin[0] + ix * 0.04f;
    float py = vol_origin[1] + iy * 0.04f;
    float pz = vol_origin[2] + iz * 0.04f;

    float sumF[C2D];
#pragma unroll
    for (int i = 0; i < C2D; ++i) sumF[i] = 0.0f;
    float cntv = 0.0f;

#pragma unroll 1
    for (int v = 0; v < NVIEW; ++v) {
        const float* kr = KR + v * 16;
        float imx = kr[0] * px + kr[1] * py + kr[2]  * pz + kr[3];
        float imy = kr[4] * px + kr[5] * py + kr[6]  * pz + kr[7];
        float imz = kr[8] * px + kr[9] * py + kr[10] * pz + kr[11];
        float zs = (fabsf(imz) < 1e-6f) ? 1e-6f : imz;
        float x = imx / zs, y = imy / zs;
        bool valid = (imz > 1e-3f) && (x >= 0.0f) && (x <= (float)(WW - 1)) &&
                     (y >= 0.0f) && (y <= (float)(HH - 1));
        float vm = valid ? 1.0f : 0.0f;
        if (!valid) { x = 0.0f; y = 0.0f; }
        float s[C2D];
        sample24(feat2d + (size_t)v * C2D * HH * WW, x, y, vm, s);
#pragma unroll
        for (int i = 0; i < C2D; ++i) sumF[i] += s[i];
        cntv += vm;
    }

    float denom = fmaxf(cntv, 1.0f);
    float nrm = 0.0f;
#pragma unroll
    for (int i = 0; i < C2D; ++i) {
        float m = sumF[i] / denom;
        sumF[i] = m;                     // reuse as mean
        nrm += m * m;
    }
    float invn = 1.0f / (sqrtf(nrm) + 1e-8f);
#pragma unroll
    for (int i = 0; i < C2D; ++i) sumF[i] *= invn;   // mn = mean / (||mean|| + eps)

    float sim = 0.0f;
#pragma unroll 1
    for (int v = 0; v < NVIEW; ++v) {
        const float* kr = KR + v * 16;
        float imx = kr[0] * px + kr[1] * py + kr[2]  * pz + kr[3];
        float imy = kr[4] * px + kr[5] * py + kr[6]  * pz + kr[7];
        float imz = kr[8] * px + kr[9] * py + kr[10] * pz + kr[11];
        float zs = (fabsf(imz) < 1e-6f) ? 1e-6f : imz;
        float x = imx / zs, y = imy / zs;
        bool valid = (imz > 1e-3f) && (x >= 0.0f) && (x <= (float)(WW - 1)) &&
                     (y >= 0.0f) && (y <= (float)(HH - 1));
        float vm = valid ? 1.0f : 0.0f;
        if (!valid) { x = 0.0f; y = 0.0f; }
        float s[C2D];
        sample24(feat2d + (size_t)v * C2D * HH * WW, x, y, vm, s);
        float dot = 0.0f, ss = 0.0f;
#pragma unroll
        for (int i = 0; i < C2D; ++i) { dot += s[i] * sumF[i]; ss += s[i] * s[i]; }
        sim += vm * dot / (sqrtf(ss) + 1e-8f);
    }
    sim /= denom;

    volBf[id] = (__bf16)sim;
    cnt[id] = cntv;
}

// ------------------------------ WMMA implicit-GEMM conv --------------------
// Block = NWAVES*32 threads, one 16-pixel n-tile per block. The block first
// gathers the implicit-GEMM B operand for ALL K-chunks into LDS, laid out in
// the exact WMMA B register order (lanes 0-15 -> K=0..7,16..23 of the chunk,
// lanes 16-31 -> K=8..15,24..31; with h = lane/16, element j -> K=8h+j (j<8), 8+8h+j (j>=8)), so each
// wave's steady state is ds_load_b128 (B) + 32B global load (A) + wmma.
// Each wave owns MT m-tiles (MT accumulators) to amortize B further.
template <int CIN, int COUT, int NUMKC, int MT, bool RELU>
__global__ void __launch_bounds__((COUT / 16 / MT) * 32)
conv_wmma(const __bf16* __restrict__ act,
          const __bf16* __restrict__ pw,
          const float* __restrict__ bias,
          __bf16* __restrict__ outBf,
          float* __restrict__ outF) {
    const int K = CIN * 9;
    const int NWAVES = COUT / 16 / MT;
    const int BDIM = NWAVES * 32;

    __shared__ __attribute__((aligned(32))) __bf16 bLds[NUMKC * 512];

    int tid = threadIdx.x;
    int nt = blockIdx.x;                 // 0..575
    int y  = nt / 6;
    int x0 = (nt - y * 6) * 16;

    // ---- phase 1: cooperative B-pack into LDS (shared by all waves) ----
    for (int idx = tid; idx < NUMKC * 512; idx += BDIM) {
        int kc = idx >> 9;
        int r  = idx & 511;
        int l  = r >> 4;                 // lane slot 0..31
        int j  = r & 15;                 // element within v16bf
        int k  = kc * 32 + ((j >= 8) ? 16 : 0) + ((l >= 16) ? 8 : 0) + (j & 7);
        int xc = x0 + (l & 15);
        __bf16 val = (__bf16)0.0f;
        if (k < K) {
            int ic = k / 9;
            int kk = k - ic * 9;
            int ky = kk / 3 - 1;
            int kx = kk - (kk / 3) * 3 - 1;
            int yy = y + ky, xx2 = xc + kx;
            if (yy >= 0 && yy < NVOX && xx2 >= 0 && xx2 < NVOX)
                val = act[ic * SPATIAL + yy * NVOX + xx2];
        }
        bLds[idx] = val;
    }
    __syncthreads();

    // ---- phase 2: per-wave WMMA over K-chunks ----
    int wave = tid >> 5;
    int lane = tid & 31;
    int n  = lane & 15;
    int xc = x0 + n;
    int hi = (lane >> 4) & 1;

    v8f acc[MT];
#pragma unroll
    for (int mi = 0; mi < MT; ++mi) acc[mi] = {};

#pragma unroll 1
    for (int kc = 0; kc < NUMKC; ++kc) {
        v16bf b = *(const v16bf*)(bLds + kc * 512 + lane * 16);   // ds_load
#pragma unroll
        for (int mi = 0; mi < MT; ++mi) {
            int mt = wave * MT + mi;
            v16bf a = *(const v16bf*)(pw + ((size_t)(mt * NUMKC + kc) * 512) + lane * 16);
            acc[mi] = __builtin_amdgcn_wmma_f32_16x16x32_bf16(
                /*neg_a=*/false, a, /*neg_b=*/false, b,
                /*c_mod=*/(short)0, acc[mi], /*reuse_a=*/false, /*reuse_b=*/false);
        }
    }

    // ---- epilogue: bias (+ReLU) and store ----
#pragma unroll
    for (int mi = 0; mi < MT; ++mi) {
        int mt = wave * MT + mi;
#pragma unroll
        for (int r2 = 0; r2 < 8; ++r2) {
            int m = mt * 16 + hi * 8 + r2;   // C/D layout: VGPR r -> M = r (+8 hi lanes)
            float v = acc[mi][r2] + bias[m];
            int oidx = m * SPATIAL + y * NVOX + xc;
            if (RELU) {
                v = fmaxf(v, 0.0f);
                outBf[oidx] = (__bf16)v;
            } else {
                outF[oidx] = v;
            }
        }
    }
}

// ------------------------------ upsample + sum -----------------------------
__global__ void upsample_sum_kernel(const float* __restrict__ out2,
                                    float* __restrict__ feat) {
    int id = blockIdx.x * blockDim.x + threadIdx.x;
    if (id >= TENSO_CH * MASK_ELEMS) return;
    int t = id / MASK_ELEMS;
    int r = id - t * MASK_ELEMS;
    int X = r / SPATIAL;
    int rem = r - X * SPATIAL;
    int Y = rem / NVOX, Z = rem - (rem / NVOX) * NVOX;

    float acc = 0.0f;
#pragma unroll
    for (int p = 0; p < 3; ++p) {
        int d, yy, xx;
        if (p == 0)      { d = X; yy = Y; xx = Z; }
        else if (p == 1) { d = Y; yy = X; xx = Z; }
        else             { d = Z; yy = X; xx = Y; }
        float s = (d + 0.5f) * 0.25f - 0.5f;   // half-pixel linear, scale 4
        float c0f = floorf(s);
        float w = s - c0f;
        int c0 = (int)c0f;
        int c1 = c0 + 1;
        c0 = (c0 < 0) ? 0 : ((c0 > 23) ? 23 : c0);
        c1 = (c1 < 0) ? 0 : ((c1 > 23) ? 23 : c1);
        const float* O = out2 + (size_t)p * (288 * SPATIAL);
        int sp = yy * NVOX + xx;
        float v0 = O[(t * 24 + c0) * SPATIAL + sp];
        float v1 = O[(t * 24 + c1) * SPATIAL + sp];
        acc += (1.0f - w) * v0 + w * v1;
    }
    feat[id] = acc;
}

// ------------------------------ grid mask ----------------------------------
__global__ void mask_kernel(const float* __restrict__ cnt, float* __restrict__ mask) {
    int id = blockIdx.x * blockDim.x + threadIdx.x;
    if (id >= MASK_ELEMS) return;
    int X = id / SPATIAL;
    int rem = id - X * SPATIAL;
    int Y = rem / NVOX, Z = rem - (rem / NVOX) * NVOX;
    bool any = false;
    any |= cnt[0 * PLANE_PTS + (X >> 2) * SPATIAL + Y * NVOX + Z] > 1.0f;
    any |= cnt[1 * PLANE_PTS + (Y >> 2) * SPATIAL + X * NVOX + Z] > 1.0f;
    any |= cnt[2 * PLANE_PTS + (Z >> 2) * SPATIAL + X * NVOX + Y] > 1.0f;
    mask[id] = any ? 1.0f : 0.0f;
}

// ------------------------------ host launch --------------------------------
extern "C" void kernel_launch(void* const* d_in, const int* in_sizes, int n_in,
                              void* d_out, int out_size, void* d_ws, size_t ws_size,
                              hipStream_t stream) {
    const float* feat2d = (const float*)d_in[0];   // (9,1,24,120,160)
    const float* KRcam  = (const float*)d_in[1];   // (9,1,4,4)
    const float* vorig  = (const float*)d_in[2];   // (1,3)
    const float* w1     = (const float*)d_in[3];   // (96,24,3,3)
    const float* b1     = (const float*)d_in[4];   // (96,)
    const float* w2     = (const float*)d_in[5];   // (288,96,3,3)
    const float* b2     = (const float*)d_in[6];   // (288,)
    float* outFeat = (float*)d_out;                // 12*96^3
    float* outMask = (float*)d_out + FEAT_ELEMS;   // 96^3

    // workspace carve-up (256B aligned)
    char* ws = (char*)d_ws;
    size_t off = 0;
    auto carve = [&](size_t bytes) {
        void* p = ws + off;
        off = (off + bytes + 255) & ~(size_t)255;
        return p;
    };
    float*  cnt   = (float*) carve((size_t)3 * PLANE_PTS * 4);        // 2.65 MB
    __bf16* volBf = (__bf16*)carve((size_t)3 * PLANE_PTS * 2);        // 1.33 MB
    __bf16* hBf   = (__bf16*)carve((size_t)3 * 96 * SPATIAL * 2);     // 5.31 MB
    float*  out2  = (float*) carve((size_t)3 * 288 * SPATIAL * 4);    // 31.85 MB
    __bf16* pw1   = (__bf16*)carve((size_t)6 * 7 * 512 * 2);          // 43 KB
    __bf16* pw2   = (__bf16*)carve((size_t)18 * 27 * 512 * 2);        // 497 KB
    (void)ws_size; (void)in_sizes; (void)n_in; (void)out_size;

    // 1) pack weights into WMMA A-layout (bf16)
    {
        int tot1 = 6 * 7 * 32;
        pack_w_kernel<<<(tot1 + 255) / 256, 256, 0, stream>>>(w1, pw1, 96, 24, 7);
        int tot2 = 18 * 27 * 32;
        pack_w_kernel<<<(tot2 + 255) / 256, 256, 0, stream>>>(w2, pw2, 288, 96, 27);
    }

    // 2) back-projection + cosine-sim for all 3 planes
    bp_kernel<<<(3 * PLANE_PTS) / 256, 256, 0, stream>>>(feat2d, KRcam, vorig, volBf, cnt);

    // 3) conv stacks per plane (WMMA implicit GEMM, LDS-staged B)
    for (int p = 0; p < 3; ++p) {
        const __bf16* act1 = volBf + (size_t)p * PLANE_PTS;
        __bf16* h = hBf + (size_t)p * 96 * SPATIAL;
        float* o2 = out2 + (size_t)p * 288 * SPATIAL;
        // conv1: 576 n-tile blocks, 6 waves x 1 m-tile, 192 threads
        conv_wmma<24, 96, 7, 1, true><<<576, 192, 0, stream>>>(act1, pw1, b1, h, nullptr);
        // conv2: 576 n-tile blocks, 6 waves x 3 m-tiles, 192 threads
        conv_wmma<96, 288, 27, 3, false><<<576, 192, 0, stream>>>(h, pw2, b2, nullptr, o2);
    }

    // 4) linear upsample along downsampled axis + sum of 3 planes
    upsample_sum_kernel<<<(TENSO_CH * MASK_ELEMS) / 256, 256, 0, stream>>>(out2, outFeat);

    // 5) nearest count resize + any(count>1)
    mask_kernel<<<MASK_ELEMS / 256, 256, 0, stream>>>(cnt, outMask);
}
